// EmbeddingCrossAttention_9947144257657
// MI455X (gfx1250) — matrix operand order repaired and checked by reading, not compiled, not run on hardware
//
#include <hip/hip_runtime.h>
#include <hip/hip_bf16.h>

// ---------------------------------------------------------------------------
// EmbeddingCrossAttention for MI455X (gfx1250, wave32, WMMA, async-LDS copy)
//
// Math: softmax over a single kv slot == 1.0 -> q/k path is dead.
//   t_ctx = (a@Wv^T+bv)@Wo^T+bo ; a_ctx = (t@Wv^T+bv)@Wo^T+bo.
// All GEMMs in bf16 via v_wmma_f32_16x16x32_bf16 (f32 accum); bias and the
// sigmoid-gate blends fused into epilogues.  Tiles are double-buffered in
// LDS using GLOBAL_LOAD_ASYNC_TO_LDS_B128 (ASYNCcnt) -> no VGPR staging.
// All tensor element indices fit in 32 bits -> unsigned offsets everywhere.
// ---------------------------------------------------------------------------

typedef __attribute__((ext_vector_type(16))) __bf16 v16bf;
typedef __attribute__((ext_vector_type(8)))  float  v8f;
typedef __attribute__((ext_vector_type(4)))  __bf16 v4bf;

#define BDIM 16384
#define EDIM 1024

#define BM 128
#define BN 128
#define BK 32
#define LDT 40   // LDS row stride in bf16 (BK + 8 pad -> 80B rows, 16B aligned)

// byte distance between buffer 0 and buffer 1 of the LDS tile set
#define BUFSTRIDE (2u * BM * LDT * 2u)   // 20480 B

union FragBF { v16bf v; uint4 u[2]; };

// ---- elementwise f32 -> bf16 (vectorized x4) ------------------------------
__global__ __launch_bounds__(256) void cvt_f32_to_bf16(const float* __restrict__ in,
                                                       __bf16* __restrict__ out, int n4) {
  int i = blockIdx.x * 256 + threadIdx.x;
  if (i >= n4) return;
  float4 f = ((const float4*)in)[i];
  v4bf o;
  o.x = (__bf16)f.x; o.y = (__bf16)f.y; o.z = (__bf16)f.z; o.w = (__bf16)f.w;
  ((v4bf*)out)[i] = o;
}

__device__ __forceinline__ unsigned lds_off(const void* p) {
  // low 32 bits of a generic pointer to __shared__ == wave-relative LDS addr
  return (unsigned)(unsigned long long)p;
}

__device__ __forceinline__ void async_ld_b128(unsigned lds_dst, unsigned voff,
                                              const __bf16* sbase) {
  asm volatile("global_load_async_to_lds_b128 %0, %1, %2"
               :: "v"(lds_dst), "v"(voff), "s"(sbase) : "memory");
}

// ---- C = A(MxK, bf16, row-major) * W(NxK, bf16, row-major)^T + bias -------
// A is virtually [A0 | A1] along K, split at Ksplit (for the gate concat);
// A0 and A1 both have row stride Ksplit elements (holds for all our calls).
// GATE: out = g*x + (1-g)*c with g = sigmoid(acc+bias); else bf16 store.
template <bool GATE>
__global__ __launch_bounds__(256) void gemm_bf16_wmma(
    const __bf16* __restrict__ A0, const __bf16* __restrict__ A1, int Ksplit,
    const __bf16* __restrict__ W, const float* __restrict__ bias,
    int M, int N, int K,
    __bf16* __restrict__ outb, float* __restrict__ outf,
    const __bf16* __restrict__ gx, const __bf16* __restrict__ gc)
{
  __shared__ __bf16 sm[2][2][BM * LDT];   // [buf][A|B][row*LDT + k], 40 KB

  const int tid  = threadIdx.x;
  const int lane = tid & 31;
  const int wave = tid >> 5;
  const int wm   = wave >> 1;      // 0..3 : 32-row strip
  const int wn   = wave & 1;       // 0..1 : 64-col strip
  const int l    = lane & 15;
  const int hi   = lane >> 4;

  const int m0 = blockIdx.y * BM;
  const int n0 = blockIdx.x * BN;

  // global->LDS tile copy: 128x32 bf16 = 512 16B chunks, 2 per thread,
  // each chunk moved by one async b128 load (no VGPR staging).
  unsigned rowA[2], rowB[2], ldsA[2], ldsB[2];
#pragma unroll
  for (int p = 0; p < 2; ++p) {
    const int c  = tid + p * 256;
    const int cr = c >> 2;           // tile row 0..127
    const int cq = (c & 3) * 8;      // k element offset 0/8/16/24
    rowA[p] = (unsigned)((m0 + cr) * Ksplit + cq) * 2u;  // byte off in A0/A1
    rowB[p] = (unsigned)((n0 + cr) * K + cq) * 2u;       // byte off in W
    ldsA[p] = lds_off(&sm[0][0][cr * LDT + cq]);
    ldsB[p] = lds_off(&sm[0][1][cr * LDT + cq]);
  }

  const int KT = K >> 5;

  auto issue = [&](int kt, int buf) {           // 4 async loads per wave
    const int kb = kt * BK;
    const __bf16* Ab;
    unsigned ka;
    if (kb < Ksplit) { Ab = A0; ka = (unsigned)(kb * 2); }
    else             { Ab = A1; ka = (unsigned)((kb - Ksplit) * 2); }
    const unsigned kbb = (unsigned)(kb * 2);
    const unsigned bo  = buf ? BUFSTRIDE : 0u;
#pragma unroll
    for (int p = 0; p < 2; ++p) {
      async_ld_b128(ldsA[p] + bo, rowA[p] + ka,  Ab);
      async_ld_b128(ldsB[p] + bo, rowB[p] + kbb, W);
    }
  };

  v8f acc[2][4];
  {
    v8f zero = {0.f, 0.f, 0.f, 0.f, 0.f, 0.f, 0.f, 0.f};
#pragma unroll
    for (int mi = 0; mi < 2; ++mi)
#pragma unroll
      for (int ni = 0; ni < 4; ++ni)
        acc[mi][ni] = zero;
  }

  issue(0, 0);                                   // tile 0 in flight

  for (int kt = 0; kt < KT; ++kt) {
    const int cur = kt & 1;
    if (kt + 1 < KT) {
      issue(kt + 1, cur ^ 1);                    // next tile in flight
      asm volatile("s_wait_asynccnt 0x4" ::: "memory");  // tile kt landed
    } else {
      asm volatile("s_wait_asynccnt 0x0" ::: "memory");
    }
    __syncthreads();                             // all waves' tile kt landed

    // ---- compute on sm[cur] ----
    const __bf16* sA = sm[cur][0];
    const __bf16* sB = sm[cur][1];
    FragBF fa[2], fb[4];
    // A 16x32 bf16 frag: lane row m holds K {hi*8..+8, 16+hi*8..+8}
#pragma unroll
    for (int mi = 0; mi < 2; ++mi) {
      const __bf16* base = sA + (wm * 32 + mi * 16 + l) * LDT;
      fa[mi].u[0] = *(const uint4*)(base + hi * 8);
      fa[mi].u[1] = *(const uint4*)(base + 16 + hi * 8);
    }
    // B 32x16 bf16 frag: lane col n holds K {hi*8..+8, 16+hi*8..+8}
#pragma unroll
    for (int ni = 0; ni < 4; ++ni) {
      const __bf16* base = sB + (wn * 64 + ni * 16 + l) * LDT;
      fb[ni].u[0] = *(const uint4*)(base + hi * 8);
      fb[ni].u[1] = *(const uint4*)(base + 16 + hi * 8);
    }
#pragma unroll
    for (int mi = 0; mi < 2; ++mi)
#pragma unroll
      for (int ni = 0; ni < 4; ++ni)
        acc[mi][ni] = __builtin_amdgcn_wmma_f32_16x16x32_bf16(
            false, fa[mi].v, false, fb[ni].v, (short)0, acc[mi][ni], false, false);

    if (kt + 1 < KT)
      __syncthreads();   // readers done before next issue() overwrites buf cur
  }

  // ---- epilogue: C/D layout: VGPR r -> row (hi*8 + r), col = l ----
  // All indices fit in 32 bits (max 16384*1024) -> saddr + u32 voffset stores.
#pragma unroll
  for (int ni = 0; ni < 4; ++ni) {
    const unsigned col  = (unsigned)(n0 + wn * 64 + ni * 16 + l);
    const float    bcol = bias[col];
#pragma unroll
    for (int mi = 0; mi < 2; ++mi) {
      const unsigned rbase = (unsigned)(m0 + wm * 32 + mi * 16 + hi * 8);
      unsigned idx = rbase * (unsigned)N + col;
#pragma unroll
      for (int r = 0; r < 8; ++r, idx += (unsigned)N) {
        const float val = acc[mi][ni][r] + bcol;
        if (GATE) {
          const float g = 1.0f / (1.0f + __expf(-val));
          const float x = (float)gx[idx];
          const float c = (float)gc[idx];
          outf[idx] = g * x + (1.0f - g) * c;
        } else {
          outb[idx] = (__bf16)val;
        }
      }
    }
  }
}

// ---------------------------------------------------------------------------
extern "C" void kernel_launch(void* const* d_in, const int* in_sizes, int n_in,
                              void* d_out, int out_size, void* d_ws, size_t ws_size,
                              hipStream_t stream) {
  const float* text  = (const float*)d_in[0];
  const float* audio = (const float*)d_in[1];
  const float* Wt  = (const float*)d_in[2];
  const float* bt  = (const float*)d_in[3];
  const float* Wa  = (const float*)d_in[4];
  const float* ba  = (const float*)d_in[5];
  // d_in[6..9] = Wq,bq,Wk,bk: dead (softmax over size-1 kv axis == 1)
  const float* Wv  = (const float*)d_in[10];
  const float* bv  = (const float*)d_in[11];
  const float* Wo  = (const float*)d_in[12];
  const float* bo  = (const float*)d_in[13];
  const float* Wgt = (const float*)d_in[14];
  const float* bgt = (const float*)d_in[15];
  const float* Wga = (const float*)d_in[16];
  const float* bga = (const float*)d_in[17];
  float* out = (float*)d_out;

  // workspace carve-up (~276 MB of bf16 buffers)
  char* p = (char*)d_ws;
  auto take = [&](size_t elems) {
    __bf16* r = (__bf16*)p;
    p += (elems * sizeof(__bf16) + 255) & ~(size_t)255;
    return r;
  };
  __bf16* tx_b  = take((size_t)BDIM * 768);
  __bf16* au_b  = take((size_t)BDIM * EDIM);
  __bf16* t_b   = take((size_t)BDIM * EDIM);
  __bf16* a_b   = take((size_t)BDIM * EDIM);
  __bf16* vt_b  = take((size_t)BDIM * EDIM);
  __bf16* va_b  = take((size_t)BDIM * EDIM);
  __bf16* tc_b  = take((size_t)BDIM * EDIM);
  __bf16* ac_b  = take((size_t)BDIM * EDIM);
  __bf16* Wt_b  = take((size_t)EDIM * 768);
  __bf16* Wa_b  = take((size_t)EDIM * EDIM);
  __bf16* Wv_b  = take((size_t)EDIM * EDIM);
  __bf16* Wo_b  = take((size_t)EDIM * EDIM);
  __bf16* Wgt_b = take((size_t)EDIM * 2048);
  __bf16* Wga_b = take((size_t)EDIM * 2048);

  auto cvt = [&](const float* s, __bf16* d, size_t n) {
    int n4 = (int)(n / 4);
    cvt_f32_to_bf16<<<dim3((n4 + 255) / 256), dim3(256), 0, stream>>>(s, d, n4);
  };
  cvt(text,  tx_b,  (size_t)BDIM * 768);
  cvt(audio, au_b,  (size_t)BDIM * EDIM);
  cvt(Wt,    Wt_b,  (size_t)EDIM * 768);
  cvt(Wa,    Wa_b,  (size_t)EDIM * EDIM);
  cvt(Wv,    Wv_b,  (size_t)EDIM * EDIM);
  cvt(Wo,    Wo_b,  (size_t)EDIM * EDIM);
  cvt(Wgt,   Wgt_b, (size_t)EDIM * 2048);
  cvt(Wga,   Wga_b, (size_t)EDIM * 2048);

  const dim3 blk(256);
  const dim3 grd(EDIM / BN, BDIM / BM);

  // t = text @ Wt^T + bt ; a = audio @ Wa^T + ba
  gemm_bf16_wmma<false><<<grd, blk, 0, stream>>>(tx_b, nullptr, 768, Wt_b, bt,
      BDIM, EDIM, 768, t_b, nullptr, nullptr, nullptr);
  gemm_bf16_wmma<false><<<grd, blk, 0, stream>>>(au_b, nullptr, 1024, Wa_b, ba,
      BDIM, EDIM, 1024, a_b, nullptr, nullptr, nullptr);
  // v_t = t @ Wv^T + bv ; v_a = a @ Wv^T + bv
  gemm_bf16_wmma<false><<<grd, blk, 0, stream>>>(t_b, nullptr, 1024, Wv_b, bv,
      BDIM, EDIM, 1024, vt_b, nullptr, nullptr, nullptr);
  gemm_bf16_wmma<false><<<grd, blk, 0, stream>>>(a_b, nullptr, 1024, Wv_b, bv,
      BDIM, EDIM, 1024, va_b, nullptr, nullptr, nullptr);
  // t_ctx = v_a @ Wo^T + bo ; a_ctx = v_t @ Wo^T + bo
  gemm_bf16_wmma<false><<<grd, blk, 0, stream>>>(va_b, nullptr, 1024, Wo_b, bo,
      BDIM, EDIM, 1024, tc_b, nullptr, nullptr, nullptr);
  gemm_bf16_wmma<false><<<grd, blk, 0, stream>>>(vt_b, nullptr, 1024, Wo_b, bo,
      BDIM, EDIM, 1024, ac_b, nullptr, nullptr, nullptr);
  // gates: g = sigmoid([x|c] @ Wg^T + bg) ; out = g*x + (1-g)*c
  gemm_bf16_wmma<true><<<grd, blk, 0, stream>>>(t_b, tc_b, 1024, Wgt_b, bgt,
      BDIM, EDIM, 2048, nullptr, out, t_b, tc_b);
  gemm_bf16_wmma<true><<<grd, blk, 0, stream>>>(a_b, ac_b, 1024, Wga_b, bga,
      BDIM, EDIM, 2048, nullptr, out + (size_t)BDIM * EDIM, a_b, ac_b);
}
